// ParallelAttention_16441134809830
// MI455X (gfx1250) — matrix operand order repaired and checked by reading, not compiled, not run on hardware
//
#include <hip/hip_runtime.h>

typedef __attribute__((ext_vector_type(16))) _Float16 v16h;
typedef __attribute__((ext_vector_type(8)))  _Float16 v8h;
typedef __attribute__((ext_vector_type(8)))  float    v8f;

#define DIMSZ    1024
#define HEADS    16
#define DIM_HEAD 64
#define INNER    (HEADS * DIM_HEAD)
#define BATCH    2
#define SEQ      2048
#define ATT_SCALE 0.125f   // 64^-0.5
#define NEG_BIG  (-1.0e30f)

static __device__ __forceinline__ v8f wmma_f16(v16h a, v16h b, v8f c) {
  // D = A(16x32 f16) * B(32x16 f16) + C(16x16 f32)
  return __builtin_amdgcn_wmma_f32_16x16x32_f16(false, a, false, b, (short)0, c,
                                                false, false);
}

static __device__ __forceinline__ float redmax16(float v) {
#pragma unroll
  for (int m = 1; m < 16; m <<= 1) v = fmaxf(v, __shfl_xor(v, m, 32));
  return v;
}
static __device__ __forceinline__ float redsum16(float v) {
#pragma unroll
  for (int m = 1; m < 16; m <<= 1) v += __shfl_xor(v, m, 32);
  return v;
}

// Issue one 16B global->LDS async copy (ASYNCcnt-tracked, CDNA5 TDM-class op).
static __device__ __forceinline__ void async_b128(unsigned lds_dst,
                                                  const void* gsrc) {
  asm volatile("global_load_async_to_lds_b128 %0, %1, off"
               :: "v"(lds_dst),
                  "v"((unsigned long long)(uintptr_t)gsrc)
               : "memory");
}
static __device__ __forceinline__ void async_b128_off16(unsigned lds_dst,
                                                        const void* gsrc) {
  asm volatile("global_load_async_to_lds_b128 %0, %1, off offset:16"
               :: "v"(lds_dst),
                  "v"((unsigned long long)(uintptr_t)gsrc)
               : "memory");
}

// ---------------------------------------------------------------------------
// Weight transpose + f32 -> f16 downconvert:  WT[n][k] = (f16) W[k][n]
// ---------------------------------------------------------------------------
__global__ void transpose_to_f16(const float* __restrict__ W,
                                 _Float16* __restrict__ WT,
                                 int Kdim, int Ncols) {
  int idx = blockIdx.x * blockDim.x + threadIdx.x;
  if (idx >= Kdim * Ncols) return;
  int n = idx / Kdim;
  int k = idx - n * Kdim;
  WT[idx] = (_Float16)W[(size_t)k * Ncols + n];
}

// padding mask (bool, True = masked) -> additive float bias {0, -1e30}
__global__ void mask_to_bias(const unsigned char* __restrict__ m,
                             float* __restrict__ pb, int n) {
  int i = blockIdx.x * blockDim.x + threadIdx.x;
  if (i < n) pb[i] = m[i] ? NEG_BIG : 0.0f;
}

// ---------------------------------------------------------------------------
// WMMA GEMM:  C[M x Ncols] = A_f32[M x Kdim] * WT_f16[Ncols x Kdim]^T
// Block: 256 threads = 8 waves, tile 64 rows x 128 cols, each wave 16x64.
// MODE 0: store f16 as qh/kh  [b][h][n][d]
// MODE 1: store f16 as vt     [b][h][d][n]   (per-head transposed V)
// MODE 2: store f32 out[row][col] = acc + bias[col]
// ---------------------------------------------------------------------------
template <int MODE>
__global__ __launch_bounds__(256) void gemm_wmma_kernel(
    const float* __restrict__ A, const _Float16* __restrict__ WT,
    const float* __restrict__ bias, float* __restrict__ outF,
    _Float16* __restrict__ outH, int Kdim, int Ncols) {
  const int lane = threadIdx.x & 31;
  const int wave = threadIdx.x >> 5;
  const int lm   = lane & 15;
  const int hf   = lane >> 4;
  const int rowBase = blockIdx.x * 64 + (wave & 3) * 16;
  const int colBase = blockIdx.y * 128 + (wave >> 2) * 64;

  v8f acc[4] = {};

  const float* arow = A + (size_t)(rowBase + lm) * Kdim;
  for (int kk = 0; kk < Kdim; kk += 32) {
    // A fragment (16x32 f16): lane = row lm; elems -> k = kk + 16*hf + e
    // (contiguous, matching B), f32->f16 in-reg
    v8f f0 = *(const v8f*)(arow + kk + 16 * hf);
    v8f f1 = *(const v8f*)(arow + kk + 16 * hf + 8);
    v16h a;
#pragma unroll
    for (int i = 0; i < 8; ++i) {
      a[i]     = (_Float16)f0[i];
      a[i + 8] = (_Float16)f1[i];
    }
#pragma unroll
    for (int t = 0; t < 4; ++t) {
      // B fragment (32x16): lane = column n, elems -> k = kk + 16*hf + e
      const _Float16* wrow =
          WT + (size_t)(colBase + t * 16 + lm) * Kdim + kk + 16 * hf;
      v16h b = *(const v16h*)wrow;
      acc[t] = wmma_f16(a, b, acc[t]);
    }
  }

#pragma unroll
  for (int t = 0; t < 4; ++t) {
#pragma unroll
    for (int r = 0; r < 8; ++r) {
      const int row = rowBase + r + 8 * hf;   // C/D layout: M = r + 8*(lane/16)
      const int col = colBase + t * 16 + lm;  //             N = lane%16
      float val = acc[t][r];
      if (MODE == 2) {
        outF[(size_t)row * Ncols + col] = val + bias[col];
      } else {
        const int bb = row >> 11;          // row / SEQ
        const int n  = row & (SEQ - 1);
        const int h  = col >> 6;           // col / DIM_HEAD
        const int d  = col & 63;
        size_t o = (MODE == 0)
                       ? ((size_t)((bb * HEADS + h) * SEQ + n) * DIM_HEAD + d)
                       : ((size_t)((bb * HEADS + h) * DIM_HEAD + d) * SEQ + n);
        outH[o] = (_Float16)val;
      }
    }
  }
}

// ---------------------------------------------------------------------------
// Causal flash attention with double-buffered async K/V^T staging in LDS.
// Grid: (SEQ/128, BATCH*HEADS), 256 threads = 8 waves; wave w owns rows
// [rowBlock + 16w, +16).  j loop: 64-column tiles, block-uniform count; waves
// past their causal range skip compute but join the prefetch + barriers.
// Obuf layout: [b][n][h*64+d] fp32 (ready for the output projection GEMM).
// ---------------------------------------------------------------------------
__global__ __launch_bounds__(256) void flash_attn_kernel(
    const _Float16* __restrict__ qh, const _Float16* __restrict__ kh,
    const _Float16* __restrict__ vt, const float* __restrict__ padbias,
    float* __restrict__ Obuf) {
  __shared__ __attribute__((aligned(32))) _Float16 Kt[2][64][64];   // [j][d]
  __shared__ __attribute__((aligned(32))) _Float16 Vtile[2][64][64]; // [d][j]
  __shared__ __attribute__((aligned(32))) _Float16 Plds[8][16][64];

  const int tid  = threadIdx.x;
  const int lane = tid & 31;
  const int wave = tid >> 5;
  const int lm   = lane & 15;
  const int hf   = lane >> 4;
  const int bh   = blockIdx.y;
  const int b    = bh >> 4;   // HEADS = 16
  const int h    = bh & 15;
  const int rowBlock = blockIdx.x * 128;
  const int rowBase  = rowBlock + wave * 16;

  const _Float16* Q   = qh + (size_t)bh * SEQ * DIM_HEAD;  // [n][d]
  const _Float16* Kg  = kh + (size_t)bh * SEQ * DIM_HEAD;  // [n][d]
  const _Float16* Vtg = vt + (size_t)bh * SEQ * DIM_HEAD;  // [d][n]
  const float* pbrow  = padbias + (size_t)b * SEQ;

  // LDS byte addresses for the async DMA destinations
  const unsigned ldsK[2] = {(unsigned)(size_t)&Kt[0][0][0],
                            (unsigned)(size_t)&Kt[1][0][0]};
  const unsigned ldsV[2] = {(unsigned)(size_t)&Vtile[0][0][0],
                            (unsigned)(size_t)&Vtile[1][0][0]};
  const int vrow = tid >> 2;          // 0..63 : d-row of the V^T tile
  const int vcq  = (tid & 3) * 16;    // 32B column chunk within the row

  // Q A-fragments for k-slices d=[0,32) and d=[32,64)
  v16h qa[2];
#pragma unroll
  for (int t = 0; t < 2; ++t) {
    const _Float16* qrow = Q + (size_t)(rowBase + lm) * DIM_HEAD + t * 32;
    v8h c0 = *(const v8h*)(qrow + 16 * hf);
    v8h c1 = *(const v8h*)(qrow + 16 * hf + 8);
    qa[t] = __builtin_shufflevector(c0, c1, 0, 1, 2, 3, 4, 5, 6, 7, 8, 9, 10,
                                    11, 12, 13, 14, 15);
  }

  v8f o[4] = {};
  float mrun[8], lrun[8];
#pragma unroll
  for (int r = 0; r < 8; ++r) { mrun[r] = NEG_BIG; lrun[r] = 0.0f; }

  const int njW = (rowBase + 16 + 63) >> 6;   // tiles this wave needs
  const int njB = (rowBlock + 128 + 63) >> 6; // tiles the block prefetches

  // ---- prefetch tile 0 (4 async B128 per thread: 8KB K + 8KB V^T) ----
  {
    const char* sk = (const char*)Kg + (size_t)tid * 32;  // jbase = 0
    async_b128(ldsK[0] + tid * 32, sk);
    async_b128_off16(ldsK[0] + tid * 32, sk);
    const char* sv = (const char*)(Vtg + (size_t)vrow * SEQ + vcq);
    async_b128(ldsV[0] + vrow * 128 + vcq * 2, sv);
    async_b128_off16(ldsV[0] + vrow * 128 + vcq * 2, sv);
  }

  for (int jt = 0; jt < njB; ++jt) {
    const int jbase = jt * 64;
    const int buf   = jt & 1;

    if (jt + 1 < njB) {  // prefetch next tile into the other buffer
      const int nb = buf ^ 1;
      const int jn = jbase + 64;
      const char* sk = (const char*)(Kg + (size_t)jn * DIM_HEAD) + tid * 32;
      async_b128(ldsK[nb] + tid * 32, sk);
      async_b128_off16(ldsK[nb] + tid * 32, sk);
      const char* sv = (const char*)(Vtg + (size_t)vrow * SEQ + jn + vcq);
      async_b128(ldsV[nb] + vrow * 128 + vcq * 2, sv);
      async_b128_off16(ldsV[nb] + vrow * 128 + vcq * 2, sv);
      asm volatile("s_wait_asynccnt 0x4" ::: "memory");  // tile jt complete
    } else {
      asm volatile("s_wait_asynccnt 0x0" ::: "memory");
    }
    __syncthreads();  // whole tile visible to all waves

    if (jt < njW) {
      // ---- S = Q * K^T : four 16-col fragments, two d-slices each ----
      v8f s0 = {}, s1 = {}, s2 = {}, s3 = {};
#pragma unroll
      for (int t = 0; t < 4; ++t) {
        const _Float16* krow = &Kt[buf][t * 16 + lm][0];
        v16h b0 = *(const v16h*)(krow + 16 * hf);        // d 0..31
        v16h b1 = *(const v16h*)(krow + 32 + 16 * hf);   // d 32..63
        v8f* st = (t == 0) ? &s0 : (t == 1) ? &s1 : (t == 2) ? &s2 : &s3;
        *st = wmma_f16(qa[0], b0, *st);
        *st = wmma_f16(qa[1], b1, *st);
      }

      // per-column padding bias (0 or -1e30) folds into the scale FMA
      const float pb0 = pbrow[jbase + lm];
      const float pb1 = pbrow[jbase + 16 + lm];
      const float pb2 = pbrow[jbase + 32 + lm];
      const float pb3 = pbrow[jbase + 48 + lm];
      const bool needMask = (jbase + 63) > rowBase;  // wave-uniform

#pragma unroll
      for (int r = 0; r < 8; ++r) {
        const int irow = rowBase + r + 8 * hf;
        float x0 = fmaf(s0[r], ATT_SCALE, pb0);
        float x1 = fmaf(s1[r], ATT_SCALE, pb1);
        float x2 = fmaf(s2[r], ATT_SCALE, pb2);
        float x3 = fmaf(s3[r], ATT_SCALE, pb3);
        if (needMask) {  // causal compares only on the diagonal tile
          if (jbase      + lm > irow) x0 = NEG_BIG;
          if (jbase + 16 + lm > irow) x1 = NEG_BIG;
          if (jbase + 32 + lm > irow) x2 = NEG_BIG;
          if (jbase + 48 + lm > irow) x3 = NEG_BIG;
        }
        float rowmax = redmax16(fmaxf(fmaxf(x0, x1), fmaxf(x2, x3)));
        float mnew = fmaxf(mrun[r], rowmax);
        float corr = __expf(mrun[r] - mnew);
        float p0 = __expf(x0 - mnew);
        float p1 = __expf(x1 - mnew);
        float p2 = __expf(x2 - mnew);
        float p3 = __expf(x3 - mnew);
        float rs = redsum16((p0 + p1) + (p2 + p3));
        lrun[r] = lrun[r] * corr + rs;
        mrun[r] = mnew;
        o[0][r] *= corr; o[1][r] *= corr; o[2][r] *= corr; o[3][r] *= corr;
        Plds[wave][r + 8 * hf][lm]      = (_Float16)p0;
        Plds[wave][r + 8 * hf][lm + 16] = (_Float16)p1;
        Plds[wave][r + 8 * hf][lm + 32] = (_Float16)p2;
        Plds[wave][r + 8 * hf][lm + 48] = (_Float16)p3;
      }

      // wave-private staging; DS is in-order per wave — drain DScnt
      asm volatile("s_wait_dscnt 0" ::: "memory");

      // reload P in A-fragment layout, two 32-k slices
      const _Float16* prow = &Plds[wave][lm][0];
      v8h c0 = *(const v8h*)(prow + 16 * hf);
      v8h c1 = *(const v8h*)(prow + 16 * hf + 8);
      v8h c2 = *(const v8h*)(prow + 32 + 16 * hf);
      v8h c3 = *(const v8h*)(prow + 40 + 16 * hf);
      v16h pa0 = __builtin_shufflevector(c0, c1, 0, 1, 2, 3, 4, 5, 6, 7, 8, 9,
                                         10, 11, 12, 13, 14, 15);
      v16h pa1 = __builtin_shufflevector(c2, c3, 0, 1, 2, 3, 4, 5, 6, 7, 8, 9,
                                         10, 11, 12, 13, 14, 15);

      // ---- O += P * V : B fragments contiguous from transposed V tile ----
#pragma unroll
      for (int t = 0; t < 4; ++t) {
        const _Float16* vrow0 = &Vtile[buf][t * 16 + lm][0];
        v16h vb0 = *(const v16h*)(vrow0 + 16 * hf);        // j 0..31
        v16h vb1 = *(const v16h*)(vrow0 + 32 + 16 * hf);   // j 32..63
        o[t] = wmma_f16(pa0, vb0, o[t]);
        o[t] = wmma_f16(pa1, vb1, o[t]);
      }
    }

    __syncthreads();  // everyone done reading before buffer reuse
  }

  // ---- normalize and store [b][n][h*64+d] fp32 ----
#pragma unroll
  for (int r = 0; r < 8; ++r) {
    const float inv = 1.0f / lrun[r];
    const int n = rowBase + r + 8 * hf;
    const size_t base = ((size_t)b * SEQ + n) * INNER + h * DIM_HEAD;
#pragma unroll
    for (int t = 0; t < 4; ++t) Obuf[base + t * 16 + lm] = o[t][r] * inv;
  }
}

// ---------------------------------------------------------------------------
extern "C" void kernel_launch(void* const* d_in, const int* in_sizes, int n_in,
                              void* d_out, int out_size, void* d_ws,
                              size_t ws_size, hipStream_t stream) {
  (void)in_sizes; (void)n_in; (void)out_size; (void)ws_size;
  const float* q  = (const float*)d_in[0];
  const float* k  = (const float*)d_in[1];
  const float* v  = (const float*)d_in[2];
  const float* Wq = (const float*)d_in[3];
  const float* Wk = (const float*)d_in[4];
  const float* Wv = (const float*)d_in[5];
  const float* Wo = (const float*)d_in[6];
  const float* bo = (const float*)d_in[7];
  const unsigned char* mask = (const unsigned char*)d_in[8];
  float* out = (float*)d_out;

  char* ws = (char*)d_ws;
  _Float16* WqT = (_Float16*)(ws + (size_t)0);          // 2 MB
  _Float16* WkT = (_Float16*)(ws + ((size_t)2  << 20)); // 2 MB
  _Float16* WvT = (_Float16*)(ws + ((size_t)4  << 20)); // 2 MB
  _Float16* WoT = (_Float16*)(ws + ((size_t)6  << 20)); // 2 MB
  _Float16* qh  = (_Float16*)(ws + ((size_t)8  << 20)); // 8 MB
  _Float16* kh  = (_Float16*)(ws + ((size_t)16 << 20)); // 8 MB
  _Float16* vt  = (_Float16*)(ws + ((size_t)24 << 20)); // 8 MB
  float*    Ob  = (float*)   (ws + ((size_t)32 << 20)); // 16 MB
  float*    pbias = (float*) (ws + ((size_t)48 << 20)); // 16 KB

  dim3 tb(256);

  const int wElems = DIMSZ * INNER;
  transpose_to_f16<<<wElems / 256, tb, 0, stream>>>(Wq, WqT, DIMSZ, INNER);
  transpose_to_f16<<<wElems / 256, tb, 0, stream>>>(Wk, WkT, DIMSZ, INNER);
  transpose_to_f16<<<wElems / 256, tb, 0, stream>>>(Wv, WvT, DIMSZ, INNER);
  transpose_to_f16<<<wElems / 256, tb, 0, stream>>>(Wo, WoT, INNER, DIMSZ);
  mask_to_bias<<<(BATCH * SEQ) / 256, tb, 0, stream>>>(mask, pbias,
                                                       BATCH * SEQ);

  dim3 gproj(BATCH * SEQ / 64, INNER / 128);
  gemm_wmma_kernel<0><<<gproj, tb, 0, stream>>>(q, WqT, nullptr, nullptr, qh,
                                                DIMSZ, INNER);
  gemm_wmma_kernel<0><<<gproj, tb, 0, stream>>>(k, WkT, nullptr, nullptr, kh,
                                                DIMSZ, INNER);
  gemm_wmma_kernel<1><<<gproj, tb, 0, stream>>>(v, WvT, nullptr, nullptr, vt,
                                                DIMSZ, INNER);

  dim3 gattn(SEQ / 128, BATCH * HEADS);
  flash_attn_kernel<<<gattn, tb, 0, stream>>>(qh, kh, vt, pbias, Ob);

  dim3 gout(BATCH * SEQ / 64, DIMSZ / 128);
  gemm_wmma_kernel<2><<<gout, tb, 0, stream>>>(Ob, WoT, bo, out, nullptr, INNER,
                                               DIMSZ);
}
